// ST_GCN_20014547599433
// MI455X (gfx1250) — matrix operand order repaired and checked by reading, not compiled, not run on hardware
//
#include <hip/hip_runtime.h>

// ---------- types ----------
typedef __attribute__((ext_vector_type(16))) __bf16 v16bf;
typedef __attribute__((ext_vector_type(8)))  __bf16 v8bf;
typedef __attribute__((ext_vector_type(8)))  float  v8f;
typedef __attribute__((ext_vector_type(4)))  unsigned uint32x4;
typedef __attribute__((ext_vector_type(8)))  int      int32x8;
typedef __attribute__((ext_vector_type(4)))  int      int32x4;

union FragBF { v16bf v; uint4 q[2]; };
union PairBF { __bf16 h[2]; unsigned u; };

__device__ __forceinline__ __bf16 f2bf(float f) {
    unsigned u = __builtin_bit_cast(unsigned, f);
    u += 0x7FFFu + ((u >> 16) & 1u);          // round-to-nearest-even
    unsigned short s = (unsigned short)(u >> 16);
    return __builtin_bit_cast(__bf16, s);
}

// Problem constants (from reference setup_inputs)
#define CIN   256
#define COUT  256
#define NPART 3
#define VDIM  2048
#define M1    768              // COUT*NPART
#define PLANE (COUT * VDIM)    // 524288 elements per (co,v) plane

// ---------------------------------------------------------------------------
// TDM: 2D bf16 tile load global->LDS with row padding matching [rows][40]bf16
// layout (row payload 64B = 16 DWORDs, then 16B pad -> stride 80B).
// D# packing per CDNA5 ISA.
// data_size=1 (2B) [17:16], pad_enable [20], pad_interval=3 (16 DW) [24:22],
// pad_amount=3 (4 DW) [31:25].
// ---------------------------------------------------------------------------
#define TDM_G1W0 ((1u << 16) | (1u << 20) | (3u << 22) | (3u << 25))

__device__ __forceinline__ void tdm_load_tile_2d(unsigned lds_byte_addr,
                                                 const void* gaddr,
                                                 unsigned tensor_d0,
                                                 unsigned tensor_d1,
                                                 unsigned stride0,
                                                 unsigned tile_d0,
                                                 unsigned tile_d1) {
    unsigned long long ga = (unsigned long long)gaddr;
    uint32x4 g0;
    g0[0] = 1u;                                        // count=1 (valid), user mode
    g0[1] = lds_byte_addr;                             // lds_addr [63:32]
    g0[2] = (unsigned)(ga & 0xFFFFFFFFu);              // global_addr lo
    g0[3] = (unsigned)((ga >> 32) & 0x01FFFFFFu) | (2u << 30);  // addr hi | type=2
    int32x8 g1;
    g1[0] = (int)TDM_G1W0;                                           // mask/ds/pad
    g1[1] = (int)((tensor_d0 & 0xFFFFu) << 16);                      // abar=0, td0 lo
    g1[2] = (int)(((tensor_d0 >> 16) & 0xFFFFu) | ((tensor_d1 & 0xFFFFu) << 16));
    g1[3] = (int)(((tensor_d1 >> 16) & 0xFFFFu) | ((tile_d0 & 0xFFFFu) << 16));
    g1[4] = (int)(tile_d1 & 0xFFFFu);                                // tile_dim2=0
    g1[5] = (int)stride0;                                            // stride0 lo
    g1[6] = 0;                                                       // stride0 hi, stride1 lo
    g1[7] = 0;
    int32x4 z4 = {0, 0, 0, 0};
    int32x8 z8 = {0, 0, 0, 0, 0, 0, 0, 0};
    __builtin_amdgcn_tensor_load_to_lds(g0, g1, z4, z4, z8, 0);
}

__device__ __forceinline__ unsigned lds_offset_of(const void* p) {
    // generic pointer to LDS: low 32 bits are the workgroup-relative LDS offset
    return (unsigned)(unsigned long long)p;
}

// ---------------------------------------------------------------------------
// Kernel 1: fp32 -> bf16 conversion of x (256x2048) and conv_w (768x256)
// ---------------------------------------------------------------------------
__global__ void convert_kernel(const float* __restrict__ x,
                               const float* __restrict__ w,
                               __bf16* __restrict__ xb,
                               __bf16* __restrict__ wb) {
    int t = blockIdx.x * blockDim.x + threadIdx.x;
    if (t < CIN * VDIM) xb[t] = f2bf(x[t]);
    int t2 = t - CIN * VDIM;
    if (t2 >= 0 && t2 < M1 * CIN) wb[t2] = f2bf(w[t2]);
}

// ---------------------------------------------------------------------------
// Kernel 2: fifo partial sum (first 16 frames x 3 partitions = 48 planes)
// ---------------------------------------------------------------------------
__global__ void fifosum_kernel(const float* __restrict__ fifo,
                               float* __restrict__ fsum) {
    size_t e = (size_t)(blockIdx.x * blockDim.x + threadIdx.x) * 4;
    float4 s = make_float4(0.f, 0.f, 0.f, 0.f);
#pragma unroll 4
    for (int j = 0; j < 48; ++j) {
        float4 v = *(const float4*)(fifo + (size_t)j * PLANE + e);
        s.x += v.x; s.y += v.y; s.z += v.z; s.w += v.w;
    }
    *(float4*)(fsum + e) = s;
}

// ---------------------------------------------------------------------------
// Kernel 3: GEMM1  Y(768x2048) = W(768x256) @ X(256x2048) + b   (bf16 WMMA)
// WG tile: M=128, N=128, full K=256. 8 waves, each 64Mx32N (4x2 accum).
// W tile staged by TDM; X tile staged transposed by VALU.
// ---------------------------------------------------------------------------
__global__ __launch_bounds__(256) void gemm1_kernel(const __bf16* __restrict__ Wb,
                                                    const __bf16* __restrict__ Xb,
                                                    const float*  __restrict__ bias,
                                                    __bf16* __restrict__ Y) {
    __shared__ __align__(16) __bf16 ldsW[128][40];   // [m][k], padded row (80B)
    __shared__ __align__(16) __bf16 ldsX[128][40];   // [n][k] (transposed)

    const int tid   = threadIdx.x;
    const int nBase = blockIdx.x * 128;
    const int mBase = blockIdx.y * 128;
    const int wave  = tid >> 5, lane = tid & 31;
    const int waveM = (wave & 1) * 64;
    const int waveN = (wave >> 1) * 32;
    const int lr = lane & 15, hs = lane >> 4;
    const unsigned ldsW_base = lds_offset_of(&ldsW[0][0]);

    v8f acc[4][2];
#pragma unroll
    for (int i = 0; i < 4; ++i)
#pragma unroll
        for (int j = 0; j < 2; ++j) acc[i][j] = (v8f){0,0,0,0,0,0,0,0};

    for (int kk = 0; kk < CIN; kk += 32) {
        // Stage W tile (128 rows x 32 bf16) via Tensor Data Mover (wave 0 only)
        if (tid < 32) {
            tdm_load_tile_2d(ldsW_base,
                             Wb + (size_t)mBase * CIN + kk,
                             /*tensor_d0=*/CIN, /*tensor_d1=*/128,
                             /*stride0=*/CIN,
                             /*tile_d0=*/32, /*tile_d1=*/128);
        }
        // Stage X transposed: all 256 threads, 2 k-rows x 8 n each
        {
            const int k0 = (tid & 15) * 2;
            const int n0 = (tid >> 4) * 8;          // 0..120
            v8bf r0 = *(const v8bf*)(Xb + (size_t)(kk + k0) * VDIM + nBase + n0);
            v8bf r1 = *(const v8bf*)(Xb + (size_t)(kk + k0 + 1) * VDIM + nBase + n0);
#pragma unroll
            for (int i = 0; i < 8; ++i) {
                PairBF pr; pr.h[0] = r0[i]; pr.h[1] = r1[i];
                *(unsigned*)&ldsX[n0 + i][k0] = pr.u;
            }
        }
        __builtin_amdgcn_s_wait_tensorcnt(0);  // wave0's TDM done (no-op elsewhere)
        __syncthreads();

        FragBF af[4], bf[2];
#pragma unroll
        for (int mt = 0; mt < 4; ++mt) {
            const __bf16* row = &ldsW[waveM + mt * 16 + lr][0];
            af[mt].q[0] = *(const uint4*)(row + hs * 8);
            af[mt].q[1] = *(const uint4*)(row + 16 + hs * 8);
        }
#pragma unroll
        for (int nt = 0; nt < 2; ++nt) {
            const __bf16* row = &ldsX[waveN + nt * 16 + lr][0];
            bf[nt].q[0] = *(const uint4*)(row + hs * 8);
            bf[nt].q[1] = *(const uint4*)(row + 16 + hs * 8);
        }
#pragma unroll
        for (int mt = 0; mt < 4; ++mt)
#pragma unroll
            for (int nt = 0; nt < 2; ++nt)
                acc[mt][nt] = __builtin_amdgcn_wmma_f32_16x16x32_bf16(
                    false, af[mt].v, false, bf[nt].v, (short)0, acc[mt][nt],
                    false, false);
        __syncthreads();
    }

    // Epilogue: add bias, convert to bf16, store Y row-major [o][v]
#pragma unroll
    for (int mt = 0; mt < 4; ++mt)
#pragma unroll
        for (int nt = 0; nt < 2; ++nt) {
            const int vcol = nBase + waveN + nt * 16 + lr;
            const int o0   = mBase + waveM + mt * 16 + hs * 8;
#pragma unroll
            for (int r = 0; r < 8; ++r) {
                float val = acc[mt][nt][r] + bias[o0 + r];
                Y[(size_t)(o0 + r) * VDIM + vcol] = f2bf(val);
            }
        }
}

// ---------------------------------------------------------------------------
// Kernel 4: GEMM2  part[p] (256x2048) = Y_p (256x2048) @ A_p (2048x2048)
// grid = (32 n-blocks, 3 partitions). WG tile: M=256, N=64, K=2048 loop.
// Y tile staged by TDM; A_p converted fp32->bf16 on the fly (transposed LDS).
// ---------------------------------------------------------------------------
__global__ __launch_bounds__(256) void gemm2_kernel(const float*  __restrict__ Amat,
                                                    const __bf16* __restrict__ Y,
                                                    float* __restrict__ part) {
    __shared__ __align__(16) __bf16 ldsY[256][40];   // [co][k], padded row (80B)
    __shared__ __align__(16) __bf16 ldsA[64][40];    // [n][k] (transposed)

    const int tid   = threadIdx.x;
    const int p     = blockIdx.y;
    const int nBase = blockIdx.x * 64;
    const int wave  = tid >> 5, lane = tid & 31;
    const int waveM = (wave & 3) * 64;
    const int waveN = (wave >> 2) * 32;
    const int lr = lane & 15, hs = lane >> 4;

    const float*  Ap = Amat + (size_t)p * VDIM * VDIM;
    const __bf16* Yp = Y    + (size_t)p * COUT * VDIM;
    const unsigned ldsY_base = lds_offset_of(&ldsY[0][0]);

    v8f acc[4][2];
#pragma unroll
    for (int i = 0; i < 4; ++i)
#pragma unroll
        for (int j = 0; j < 2; ++j) acc[i][j] = (v8f){0,0,0,0,0,0,0,0};

    for (int kk = 0; kk < VDIM; kk += 32) {
        // Stage Y tile (256 rows x 32 bf16) via Tensor Data Mover (wave 0 only)
        if (tid < 32) {
            tdm_load_tile_2d(ldsY_base,
                             Yp + kk,
                             /*tensor_d0=*/VDIM, /*tensor_d1=*/COUT,
                             /*stride0=*/VDIM,
                             /*tile_d0=*/32, /*tile_d1=*/COUT);
        }
        // Stage A_p transposed + bf16 convert (wave-uniform branch: waves 0-3)
        if (tid < 128) {
            const int k0 = (tid & 15) * 2;
            const int n0 = (tid >> 4) * 8;          // 0..56
            const float* r0 = Ap + (size_t)(kk + k0) * VDIM + nBase + n0;
            const float* r1 = r0 + VDIM;
            if (kk + 32 < VDIM) __builtin_prefetch(r0 + 32 * VDIM, 0, 0);
            float4 f0 = ((const float4*)r0)[0], f1 = ((const float4*)r0)[1];
            float4 g0 = ((const float4*)r1)[0], g1 = ((const float4*)r1)[1];
            float a0[8] = {f0.x, f0.y, f0.z, f0.w, f1.x, f1.y, f1.z, f1.w};
            float a1[8] = {g0.x, g0.y, g0.z, g0.w, g1.x, g1.y, g1.z, g1.w};
#pragma unroll
            for (int i = 0; i < 8; ++i) {
                PairBF pr; pr.h[0] = f2bf(a0[i]); pr.h[1] = f2bf(a1[i]);
                *(unsigned*)&ldsA[n0 + i][k0] = pr.u;
            }
        }
        __builtin_amdgcn_s_wait_tensorcnt(0);  // wave0's TDM done (no-op elsewhere)
        __syncthreads();

        FragBF af[4], bf[2];
#pragma unroll
        for (int mt = 0; mt < 4; ++mt) {
            const __bf16* row = &ldsY[waveM + mt * 16 + lr][0];
            af[mt].q[0] = *(const uint4*)(row + hs * 8);
            af[mt].q[1] = *(const uint4*)(row + 16 + hs * 8);
        }
#pragma unroll
        for (int nt = 0; nt < 2; ++nt) {
            const __bf16* row = &ldsA[waveN + nt * 16 + lr][0];
            bf[nt].q[0] = *(const uint4*)(row + hs * 8);
            bf[nt].q[1] = *(const uint4*)(row + 16 + hs * 8);
        }
#pragma unroll
        for (int mt = 0; mt < 4; ++mt)
#pragma unroll
            for (int nt = 0; nt < 2; ++nt)
                acc[mt][nt] = __builtin_amdgcn_wmma_f32_16x16x32_bf16(
                    false, af[mt].v, false, bf[nt].v, (short)0, acc[mt][nt],
                    false, false);
        __syncthreads();
    }

    // Write fp32 partial for this partition (deterministic, no atomics)
    float* outp = part + (size_t)p * PLANE;
#pragma unroll
    for (int mt = 0; mt < 4; ++mt)
#pragma unroll
        for (int nt = 0; nt < 2; ++nt) {
            const int vcol = nBase + waveN + nt * 16 + lr;
            const int co0  = waveM + mt * 16 + hs * 8;
#pragma unroll
            for (int r = 0; r < 8; ++r)
                outp[(size_t)(co0 + r) * VDIM + vcol] = acc[mt][nt][r];
        }
}

// ---------------------------------------------------------------------------
// Kernel 5: e = part0+part1+part2+fsum ; out[v][co] = relu(e[co][v])
// 32x32 LDS transpose tiles.
// ---------------------------------------------------------------------------
__global__ __launch_bounds__(256) void epilogue_kernel(const float* __restrict__ part,
                                                       const float* __restrict__ fsum,
                                                       float* __restrict__ out) {
    __shared__ float tile[32][33];
    const int coBase = blockIdx.x * 32;    // 8 blocks over C_out
    const int vBase  = blockIdx.y * 32;    // 64 blocks over V
    const int tx = threadIdx.x & 31;
    const int ty = threadIdx.x >> 5;       // 0..7
#pragma unroll
    for (int r = 0; r < 32; r += 8) {
        const int co = coBase + ty + r;
        const int v  = vBase + tx;
        const size_t idx = (size_t)co * VDIM + v;
        float s = fsum[idx] + part[idx] + part[(size_t)PLANE + idx] +
                  part[2 * (size_t)PLANE + idx];
        tile[ty + r][tx] = s;
    }
    __syncthreads();
#pragma unroll
    for (int r = 0; r < 32; r += 8) {
        const int v  = vBase + ty + r;
        const int co = coBase + tx;
        out[(size_t)v * COUT + co] = fmaxf(tile[tx][ty + r], 0.f);
    }
}

// ---------------------------------------------------------------------------
// Host launcher
// ---------------------------------------------------------------------------
extern "C" void kernel_launch(void* const* d_in, const int* in_sizes, int n_in,
                              void* d_out, int out_size, void* d_ws, size_t ws_size,
                              hipStream_t stream) {
    (void)in_sizes; (void)n_in; (void)out_size; (void)ws_size;
    const float* x      = (const float*)d_in[0];   // (1,256,2048,1)
    const float* A      = (const float*)d_in[1];   // (3,2048,2048)
    const float* fifo   = (const float*)d_in[2];   // (17,3,256,2048)
    const float* conv_w = (const float*)d_in[3];   // (768,256,1,1)
    const float* conv_b = (const float*)d_in[4];   // (768,)
    float* out = (float*)d_out;                    // (1,2048,256)

    char* ws = (char*)d_ws;
    __bf16* ws_x    = (__bf16*)(ws + 0);          //  256*2048 bf16 = 1.0 MB
    __bf16* ws_w    = (__bf16*)(ws + 1048576);    //  768*256  bf16 = 0.375 MB
    __bf16* ws_y    = (__bf16*)(ws + 1441792);    //  768*2048 bf16 = 3.0 MB
    float*  ws_fsum = (float*) (ws + 4587520);    //  256*2048 f32  = 2.0 MB
    float*  ws_part = (float*) (ws + 6684672);    //  3*256*2048 f32 = 6.0 MB
                                                  //  total ~12.4 MB

    convert_kernel <<<dim3((CIN * VDIM + M1 * CIN) / 256), 256, 0, stream>>>(x, conv_w, ws_x, ws_w);
    fifosum_kernel <<<dim3(PLANE / 4 / 256), 256, 0, stream>>>(fifo, ws_fsum);
    gemm1_kernel   <<<dim3(VDIM / 128, M1 / 128), 256, 0, stream>>>(ws_w, ws_x, conv_b, ws_y);
    gemm2_kernel   <<<dim3(VDIM / 64, NPART), 256, 0, stream>>>(A, ws_y, ws_part);
    epilogue_kernel<<<dim3(COUT / 32, VDIM / 32), 256, 0, stream>>>(ws_part, ws_fsum, out);
}
